// GAT_29291676959273
// MI455X (gfx1250) — compile-verified, matrix operand order checked by reading, not run on hardware
//
#include <hip/hip_runtime.h>
#include <hip/hip_bf16.h>
#include <math.h>

typedef __attribute__((ext_vector_type(16))) _Float16 v16h;
typedef __attribute__((ext_vector_type(8)))  float    v8f;

#define N_NODES 100000
#define N_EDGES 1000000
#define NH 4          // heads
#define NF 64         // features per head
#define HF 256        // NH*NF (output cols of GEMM)
#define KD 64         // inner dim (D_IN == F == 64)

// ---------------------------------------------------------------------------
// Pre-pack W[64,256] f32 -> f16 fragments in WMMA B-operand order.
// Fragment index fi = colTile*2 + kstep (32 fragments); each fragment is
// 32 lanes x 16 f16 contiguous, so the GEMM loads one v16h (32B) per frag.
// ISA 16-bit B layout: lane = column (mod 16), lane>>4 selects K subgroup;
// element i<8 -> K = ks*32 + half*8 + i ; i>=8 -> K = ks*32 + 16 + half*8 + i-8
// ---------------------------------------------------------------------------
__global__ __launch_bounds__(256) void gat_pack_w(const float* __restrict__ W,
                                                  _Float16* __restrict__ wp)
{
  const int tid = blockIdx.x * blockDim.x + threadIdx.x;   // 0..1023
  const int lane = tid & 31;
  const int fi   = tid >> 5;       // colTile*2 + ks
  const int ks   = fi & 1;
  const int T    = fi >> 1;        // column tile 0..15
  const int col  = T * 16 + (lane & 15);
  const int half = lane >> 4;
  v16h b;
  #pragma unroll
  for (int i = 0; i < 8; ++i) {
    b[i]     = (_Float16)W[(size_t)(ks * 32      + half * 8 + i) * HF + col];
    b[8 + i] = (_Float16)W[(size_t)(ks * 32 + 16 + half * 8 + i) * HF + col];
  }
  *(v16h*)(wp + (size_t)tid * 16) = b;
}

// ---------------------------------------------------------------------------
// Z[N,256] = X[N,64] @ W[64,256], f16 WMMA / f32 accumulate.
// Block = 128 threads = 4 waves, covering 32 rows x 256 cols.
// Each wave owns a 64-col slab; 2 row-tiles share each B fragment.
// A fragments staged to LDS in fragment order (f16) during load.
// N_NODES % 32 == 0 -> no bounds guards needed.
// ---------------------------------------------------------------------------
__global__ __launch_bounds__(128) void gat_gemm_wmma(
    const float* __restrict__ X, const _Float16* __restrict__ wp,
    float* __restrict__ Z)
{
  __shared__ _Float16 afrag[2][2][32][16];   // [rowtile][kstep][lane][elem]
  const int tid = threadIdx.x;
  const int m0  = blockIdx.x * 32;

  {
    // 128 threads == 2 rowtiles x 2 ksteps x 32 lanes: each thread builds
    // one lane's 16 f16 A-fragment elements and stores them contiguously.
    const int rt   = tid >> 6;
    const int ks   = (tid >> 5) & 1;
    const int lane = tid & 31;
    const int row  = m0 + rt * 16 + (lane & 15);
    const int half = lane >> 4;
    const float* xr = X + (size_t)row * KD + ks * 32 + half * 8;
    v16h a;
    #pragma unroll
    for (int i = 0; i < 8; ++i) {
      a[i]     = (_Float16)xr[i];
      a[8 + i] = (_Float16)xr[16 + i];
    }
    *(v16h*)(&afrag[rt][ks][lane][0]) = a;
  }
  __syncthreads();

  const int lane = tid & 31;
  const int slab = tid >> 5;      // 0..3 -> column slab [slab*64, slab*64+64)
  const int half = lane >> 4;
  const int lr   = lane & 15;

  const v16h a00 = *(const v16h*)(&afrag[0][0][lane][0]);
  const v16h a01 = *(const v16h*)(&afrag[0][1][lane][0]);
  const v16h a10 = *(const v16h*)(&afrag[1][0][lane][0]);
  const v16h a11 = *(const v16h*)(&afrag[1][1][lane][0]);

  #pragma unroll
  for (int t = 0; t < 4; ++t) {
    const int T = slab * 4 + t;
    const v16h b0 = *(const v16h*)(wp + ((size_t)(T * 2 + 0) * 32 + lane) * 16);
    const v16h b1 = *(const v16h*)(wp + ((size_t)(T * 2 + 1) * 32 + lane) * 16);
    v8f c0 = {}, c1 = {};
    c0 = __builtin_amdgcn_wmma_f32_16x16x32_f16(false, a00, false, b0,
                                                (short)0, c0, false, false);
    c0 = __builtin_amdgcn_wmma_f32_16x16x32_f16(false, a01, false, b1,
                                                (short)0, c0, false, false);
    c1 = __builtin_amdgcn_wmma_f32_16x16x32_f16(false, a10, false, b0,
                                                (short)0, c1, false, false);
    c1 = __builtin_amdgcn_wmma_f32_16x16x32_f16(false, a11, false, b1,
                                                (short)0, c1, false, false);
    // D layout: lane half selects M subgroup; VGPR v = row offset.
    const int col = T * 16 + lr;
    float* z0 = Z + (size_t)(m0 + half * 8) * HF + col;
    float* z1 = z0 + 16 * HF;
    #pragma unroll
    for (int v = 0; v < 8; ++v) {
      z0[v * HF] = c0[v];
      z1[v * HF] = c1[v];
    }
  }
}

// ---------------------------------------------------------------------------
// el[n,h] = sum_f z[n,h,f]*al[h,f] ; er likewise. One thread per (n,h).
// ---------------------------------------------------------------------------
__global__ __launch_bounds__(256) void gat_scores(
    const float* __restrict__ Z, const float* __restrict__ al,
    const float* __restrict__ ar, float* __restrict__ el,
    float* __restrict__ er)
{
  const int id = blockIdx.x * blockDim.x + threadIdx.x;
  if (id >= N_NODES * NH) return;
  const int n = id >> 2, h = id & 3;
  const float4* zp  = (const float4*)(Z + (size_t)n * HF + h * NF);
  const float4* alp = (const float4*)(al + h * NF);
  const float4* arp = (const float4*)(ar + h * NF);
  float sl = 0.f, sr = 0.f;
  #pragma unroll
  for (int q = 0; q < NF / 4; ++q) {
    const float4 z4 = zp[q], a4 = alp[q], r4 = arp[q];
    sl += z4.x * a4.x + z4.y * a4.y + z4.z * a4.z + z4.w * a4.w;
    sr += z4.x * r4.x + z4.y * r4.y + z4.z * r4.z + z4.w * r4.w;
  }
  el[id] = sl;
  er[id] = sr;
}

__global__ __launch_bounds__(256) void gat_init_ms(float* __restrict__ mx,
                                                  float* __restrict__ sm)
{
  const int id = blockIdx.x * blockDim.x + threadIdx.x;
  if (id >= N_NODES * NH) return;
  mx[id] = -3.0e38f;
  sm[id] = 0.0f;
}

// Pass 1: e = leaky_relu(el[src]+er[dst]); store; atomic max into m[dst].
__global__ __launch_bounds__(256) void gat_edge_max(
    const int* __restrict__ src, const int* __restrict__ dst,
    const float* __restrict__ el, const float* __restrict__ er,
    float* __restrict__ ebuf, float* __restrict__ mx)
{
  const int e = blockIdx.x * blockDim.x + threadIdx.x;
  if (e >= N_EDGES) return;
  const int s = src[e], d = dst[e];
  const float4 l = ((const float4*)el)[s];
  const float4 r = ((const float4*)er)[d];
  float v[4] = {l.x + r.x, l.y + r.y, l.z + r.z, l.w + r.w};
  #pragma unroll
  for (int h = 0; h < NH; ++h) {
    v[h] = (v[h] > 0.f) ? v[h] : 0.2f * v[h];
    atomicMax(&mx[d * NH + h], v[h]);
  }
  ((float4*)ebuf)[e] = make_float4(v[0], v[1], v[2], v[3]);
}

// Pass 2: ex = exp(e - m[dst]); store back; atomic add into s[dst].
__global__ __launch_bounds__(256) void gat_edge_expsum(
    const int* __restrict__ dst, float* __restrict__ ebuf,
    const float* __restrict__ mx, float* __restrict__ sm)
{
  const int e = blockIdx.x * blockDim.x + threadIdx.x;
  if (e >= N_EDGES) return;
  const int d = dst[e];
  const float4 ev = ((const float4*)ebuf)[e];
  const float4 m  = ((const float4*)mx)[d];
  float ex[4] = {__expf(ev.x - m.x), __expf(ev.y - m.y),
                 __expf(ev.z - m.z), __expf(ev.w - m.w)};
  #pragma unroll
  for (int h = 0; h < NH; ++h) atomicAdd(&sm[d * NH + h], ex[h]);
  ((float4*)ebuf)[e] = make_float4(ex[0], ex[1], ex[2], ex[3]);
}

// Pass 3: agg[dst] += z[src] * alpha. One wave per edge; lane j handles
// columns j, j+32, ..., j+224 -> fully coalesced z reads & atomic adds.
__global__ __launch_bounds__(256) void gat_edge_scatter(
    const int* __restrict__ src, const int* __restrict__ dst,
    const float* __restrict__ ebuf, const float* __restrict__ sm,
    const float* __restrict__ Z, float* __restrict__ agg)
{
  const int lane = threadIdx.x & 31;
  const int e = blockIdx.x * 8 + (threadIdx.x >> 5);
  const int s = src[e], d = dst[e];
  const float4 ex = ((const float4*)ebuf)[e];
  const float4 sv = ((const float4*)sm)[d];
  const float alpha[4] = {ex.x / sv.x, ex.y / sv.y, ex.z / sv.z, ex.w / sv.w};
  const float* zp = Z + (size_t)s * HF;
  float* ap = agg + (size_t)d * HF;
  #pragma unroll
  for (int j = 0; j < 8; ++j) {
    const int col = j * 32 + lane;
    atomicAdd(&ap[col], zp[col] * alpha[col >> 6]);
  }
}

// out[n,f] = mean_h act(agg[n,h,f] + b[h,f]); grid is exact (N*NF % 256 == 0)
__global__ __launch_bounds__(256) void gat_finalize(
    const float* __restrict__ agg, const float* __restrict__ b,
    float* __restrict__ out, int use_tanh)
{
  const int id = blockIdx.x * blockDim.x + threadIdx.x;
  const int n = id >> 6, f = id & 63;
  const float* ap = agg + (size_t)n * HF + f;
  float acc = 0.f;
  #pragma unroll
  for (int h = 0; h < NH; ++h) {
    float v = ap[h * NF] + b[h * NF + f];
    acc += use_tanh ? tanhf(v) : v;
  }
  out[id] = acc * 0.25f;
}

extern "C" void kernel_launch(void* const* d_in, const int* in_sizes, int n_in,
                              void* d_out, int out_size, void* d_ws, size_t ws_size,
                              hipStream_t stream) {
  (void)in_sizes; (void)n_in; (void)out_size; (void)ws_size;
  const float* x   = (const float*)d_in[0];
  const int*   src = (const int*)d_in[1];
  const int*   dst = (const int*)d_in[2];
  const float* W1  = (const float*)d_in[3];
  const float* al1 = (const float*)d_in[4];
  const float* ar1 = (const float*)d_in[5];
  const float* b1  = (const float*)d_in[6];
  const float* W2  = (const float*)d_in[7];
  const float* al2 = (const float*)d_in[8];
  const float* ar2 = (const float*)d_in[9];
  const float* b2  = (const float*)d_in[10];

  float* ws   = (float*)d_ws;
  float* z    = ws;                                    // N*256 floats
  float* agg  = z    + (size_t)N_NODES * HF;           // N*256
  float* el   = agg  + (size_t)N_NODES * HF;           // N*4
  float* er   = el   + (size_t)N_NODES * NH;           // N*4
  float* mx   = er   + (size_t)N_NODES * NH;           // N*4
  float* sm   = mx   + (size_t)N_NODES * NH;           // N*4
  float* ebuf = sm   + (size_t)N_NODES * NH;           // E*4
  float* h1   = ebuf + (size_t)N_EDGES * NH;           // N*64
  _Float16* wpack = (_Float16*)(h1 + (size_t)N_NODES * NF);  // 16384 f16

  const int gGemm  = N_NODES / 32;                     // 3125 (exact)
  const int gNH    = (N_NODES * NH + 255) / 256;
  const int gE     = (N_EDGES + 255) / 256;
  const int gScat  = N_EDGES / 8;                      // exact
  const int gNF    = N_NODES * NF / 256;               // exact

  auto layer = [&](const float* xin, const float* W, const float* al,
                   const float* ar, const float* b, float* out, int use_tanh) {
    gat_pack_w<<<4, 256, 0, stream>>>(W, wpack);
    gat_gemm_wmma<<<gGemm, 128, 0, stream>>>(xin, wpack, z);
    gat_scores<<<gNH, 256, 0, stream>>>(z, al, ar, el, er);
    gat_init_ms<<<gNH, 256, 0, stream>>>(mx, sm);
    hipMemsetAsync(agg, 0, sizeof(float) * (size_t)N_NODES * HF, stream);
    gat_edge_max<<<gE, 256, 0, stream>>>(src, dst, el, er, ebuf, mx);
    gat_edge_expsum<<<gE, 256, 0, stream>>>(dst, ebuf, mx, sm);
    gat_edge_scatter<<<gScat, 256, 0, stream>>>(src, dst, ebuf, sm, z, agg);
    gat_finalize<<<gNF, 256, 0, stream>>>(agg, b, out, use_tanh);
  };

  layer(x,  W1, al1, ar1, b1, h1, 1);              // layer 1: tanh
  layer(h1, W2, al2, ar2, b2, (float*)d_out, 0);   // layer 2: identity
}
